// QuantizedViTAttention_87814901334016
// MI455X (gfx1250) — compile-verified, matrix operand order checked by reading, not run on hardware
//
#include <hip/hip_runtime.h>
#include <hip/hip_bf16.h>
#include <stdint.h>


typedef __attribute__((ext_vector_type(2)))  int      v2i;
typedef __attribute__((ext_vector_type(8)))  int      v8i;
typedef __attribute__((ext_vector_type(8)))  float    v8f;
typedef __attribute__((ext_vector_type(8)))  _Float16 v8h;
typedef __attribute__((ext_vector_type(16))) _Float16 v16h;

#define B_DIM 64
#define S_LEN 197
#define D_DIM 768
#define H_NUM 12
#define HD    64
#define ROWS  (B_DIM * S_LEN)
#define ROWT  (ROWS / 16)
#define CGRP  (D_DIM / 64)
#define SPAD  224

#if __has_builtin(__builtin_amdgcn_global_load_async_to_lds_b128) && \
    __has_builtin(__builtin_amdgcn_s_wait_asynccnt)
#define HAVE_ASYNC_LDS 1
typedef int b128_t __attribute__((vector_size(16)));
typedef __attribute__((address_space(1))) b128_t* gptr_b128;
typedef __attribute__((address_space(3))) b128_t* lptr_b128;
#else
#define HAVE_ASYNC_LDS 0
#endif

__global__ void absmax_kernel(const float* __restrict__ x, size_t n,
                              unsigned int* __restrict__ amax) {
    __shared__ unsigned int red[256];
    float m = 0.0f;
    for (size_t i = (size_t)blockIdx.x * blockDim.x + threadIdx.x; i < n;
         i += (size_t)gridDim.x * blockDim.x)
        m = fmaxf(m, fabsf(x[i]));
    red[threadIdx.x] = __float_as_uint(m);
    __syncthreads();
    for (int s = 128; s > 0; s >>= 1) {
        if ((int)threadIdx.x < s)
            red[threadIdx.x] = max(red[threadIdx.x], red[threadIdx.x + s]);
        __syncthreads();
    }
    if (threadIdx.x == 0) atomicMax(amax, red[0]);
}

__global__ void quant_w_kernel(const float* __restrict__ Wq, const float* __restrict__ bq,
                               const float* __restrict__ Wk, const float* __restrict__ bk,
                               const float* __restrict__ Wv, const float* __restrict__ bv,
                               const float* __restrict__ amax,
                               int8_t* __restrict__ w8,
                               float* __restrict__ bias_scale,
                               float* __restrict__ b_int) {
    int idx = blockIdx.x * blockDim.x + threadIdx.x;
    if (idx >= 3 * D_DIM) return;
    int mat = idx / D_DIM, o = idx % D_DIM;
    const float* W    = (mat == 0) ? Wq : (mat == 1) ? Wk : Wv;
    const float* bias = (mat == 0) ? bq : (mat == 1) ? bk : bv;
    float sf = amax[0] * (1.0f / 127.0f);
    const float* wr = W + (size_t)o * D_DIM;
    float wmax = 0.0f;
    for (int k = 0; k < D_DIM; ++k) wmax = fmaxf(wmax, fabsf(wr[k]));
    float wscale = wmax * (1.0f / 127.0f);
    int8_t* wout = w8 + (size_t)mat * D_DIM * D_DIM + (size_t)o * D_DIM;
    float inv_ws = 1.0f / wscale;
    for (int k = 0; k < D_DIM; ++k) {
        float q = rintf(wr[k] * inv_ws);
        q = fminf(fmaxf(q, -128.0f), 127.0f);
        wout[k] = (int8_t)q;
    }
    float bscale = wscale * sf;
    bias_scale[idx] = bscale;
    b_int[idx] = rintf(bias[o] / bscale);
}

__global__ void quant_x_kernel(const float* __restrict__ x,
                               const float* __restrict__ amax,
                               int8_t* __restrict__ x8, size_t n) {
    float inv_sf = 127.0f / amax[0];
    for (size_t i = (size_t)blockIdx.x * blockDim.x + threadIdx.x; i < n;
         i += (size_t)gridDim.x * blockDim.x) {
        float q = rintf(x[i] * inv_sf);
        q = fminf(fmaxf(q, -128.0f), 127.0f);
        x8[i] = (int8_t)q;
    }
}

__device__ __forceinline__ v8i load_frag8(const int8_t* p) {
    v2i c0 = *(const v2i*)(p);
    v2i c1 = *(const v2i*)(p + 16);
    v2i c2 = *(const v2i*)(p + 32);
    v2i c3 = *(const v2i*)(p + 48);
    return (v8i){c0.x, c0.y, c1.x, c1.y, c2.x, c2.y, c3.x, c3.y};
}

__device__ __forceinline__ v16h load_frag16(const _Float16* p) {
    v8h lo = *(const v8h*)(p);
    v8h hi = *(const v8h*)(p + 16);
    return __builtin_shufflevector(lo, hi, 0, 1, 2, 3, 4, 5, 6, 7,
                                   8, 9, 10, 11, 12, 13, 14, 15);
}

__global__ __launch_bounds__(128)
void qkv_wmma_kernel(const int8_t* __restrict__ x8,
                     const int8_t* __restrict__ w8,
                     const float* __restrict__ bias_scale,
                     const float* __restrict__ b_int,
                     _Float16* __restrict__ Qh,
                     _Float16* __restrict__ Kh,
                     _Float16* __restrict__ Vt) {
    const int lane = threadIdx.x & 31;
    const int wave = threadIdx.x >> 5;
    const int wid  = blockIdx.x * 4 + wave;
    const int tilesPerMat = ROWT * CGRP;
    const int mat = wid / tilesPerMat;
    const int rem = wid % tilesPerMat;
    const int rt = rem / CGRP, cg = rem % CGRP;

    const int half = lane >> 4, mn = lane & 15;
    const int8_t* xr = x8 + (size_t)(rt * 16 + mn) * D_DIM + 8 * half;
    const int8_t* wbase = w8 + (size_t)mat * D_DIM * D_DIM + 8 * half;
    const int8_t* wr0 = wbase + (size_t)(cg * 64 + 0 * 16 + mn) * D_DIM;
    const int8_t* wr1 = wbase + (size_t)(cg * 64 + 1 * 16 + mn) * D_DIM;
    const int8_t* wr2 = wbase + (size_t)(cg * 64 + 2 * 16 + mn) * D_DIM;
    const int8_t* wr3 = wbase + (size_t)(cg * 64 + 3 * 16 + mn) * D_DIM;

    v8i acc0 = {}, acc1 = {}, acc2 = {}, acc3 = {};
    for (int k0 = 0; k0 < D_DIM; k0 += 64) {
        __builtin_prefetch((const void*)(xr + k0 + 256), 0, 0);
        const v8i a = load_frag8(xr + k0);
        const v8i b0 = load_frag8(wr0 + k0);
        const v8i b1 = load_frag8(wr1 + k0);
        const v8i b2 = load_frag8(wr2 + k0);
        const v8i b3 = load_frag8(wr3 + k0);
        acc0 = __builtin_amdgcn_wmma_i32_16x16x64_iu8(true, a, true, b0, acc0, false, false);
        acc1 = __builtin_amdgcn_wmma_i32_16x16x64_iu8(true, a, true, b1, acc1, false, false);
        acc2 = __builtin_amdgcn_wmma_i32_16x16x64_iu8(true, a, true, b2, acc2, false, false);
        acc3 = __builtin_amdgcn_wmma_i32_16x16x64_iu8(true, a, true, b3, acc3, false, false);
    }

    v8i accs[4] = {acc0, acc1, acc2, acc3};
#pragma unroll
    for (int nt = 0; nt < 4; ++nt) {
        const int col = cg * 64 + nt * 16 + mn;
        const float bs = bias_scale[mat * D_DIM + col];
        const float bi = b_int[mat * D_DIM + col];
#pragma unroll
        for (int r = 0; r < 8; ++r) {
            const int orow = rt * 16 + r + 8 * half;
            const _Float16 val = (_Float16)(((float)accs[nt][r] + bi) * bs);
            if (mat == 0) {
                Qh[(size_t)orow * D_DIM + col] = val;
            } else if (mat == 1) {
                Kh[(size_t)orow * D_DIM + col] = val;
            } else {
                const int bb = orow / S_LEN, ss = orow % S_LEN;
                Vt[((size_t)(bb * H_NUM + cg) * HD + nt * 16 + mn) * SPAD + ss] = val;
            }
        }
    }
}

__global__ __launch_bounds__(128)
void attn_wmma_kernel(const _Float16* __restrict__ Qh,
                      const _Float16* __restrict__ Kh,
                      const _Float16* __restrict__ Vt,
                      float* __restrict__ out,
                      const float* __restrict__ amax) {
    __shared__ __align__(16) _Float16 vtile[HD][SPAD];
    __shared__ __align__(16) _Float16 probs[4][16][SPAD];

    const int lane = threadIdx.x & 31;
    const int wave = threadIdx.x >> 5;
    const int bh = blockIdx.x >> 2;
    const int b = bh / H_NUM, h = bh % H_NUM;
    int tile = (blockIdx.x & 3) * 4 + wave;
    const bool active = tile < 13;
    if (!active) tile = 12;

    const float sf = amax[0] * (1.0f / 127.0f);
    const int half = lane >> 4, mn = lane & 15;
    const size_t baseRow = (size_t)b * S_LEN;
    const size_t headOff = (size_t)h * HD;

#if HAVE_ASYNC_LDS
    {
        const char* srcb = (const char*)(Vt + (size_t)bh * HD * SPAD);
        char* dstb = (char*)&vtile[0][0];
#pragma unroll
        for (int i = 0; i < (HD * SPAD * 2 / 16) / 128; ++i) {
            const size_t byteOff = ((size_t)threadIdx.x + i * 128) * 16;
            __builtin_amdgcn_global_load_async_to_lds_b128(
                (gptr_b128)(uintptr_t)(srcb + byteOff),
                (lptr_b128)(uintptr_t)(dstb + byteOff),
                0, 0);
        }
        __builtin_amdgcn_s_wait_asynccnt(0);
    }
#else
    {
        const uint4* src = (const uint4*)(Vt + (size_t)bh * HD * SPAD);
        uint4* dst = (uint4*)&vtile[0][0];
#pragma unroll
        for (int i = 0; i < (HD * SPAD / 8) / 128; ++i)
            dst[threadIdx.x + i * 128] = src[threadIdx.x + i * 128];
    }
#endif
    __syncthreads();

    int qrow = tile * 16 + mn;
    if (qrow > S_LEN - 1) qrow = S_LEN - 1;
    const _Float16* qp = Qh + (baseRow + qrow) * D_DIM + headOff + 8 * half;
    const v16h a0 = load_frag16(qp);
    const v16h a1 = load_frag16(qp + 32);

    v8f s[13];
#pragma unroll
    for (int j = 0; j < 13; ++j) {
        int krow = j * 16 + mn;
        if (krow > S_LEN - 1) krow = S_LEN - 1;
        const _Float16* kp = Kh + (baseRow + krow) * D_DIM + headOff + 8 * half;
        const v16h b0 = load_frag16(kp);
        const v16h b1 = load_frag16(kp + 32);
        v8f c = {};
        c = __builtin_amdgcn_wmma_f32_16x16x32_f16(false, a0, false, b0,
                                                   (short)0, c, false, false);
        c = __builtin_amdgcn_wmma_f32_16x16x32_f16(false, a1, false, b1,
                                                   (short)0, c, false, false);
        s[j] = c * 0.125f;
    }

    const float inv_sf = 1.0f / sf;
    const float x0i = floorf(-0.6931f * inv_sf);
    const float bco = floorf((0.96963238f / 0.35815147f) * inv_sf);
    const float cco = floorf((1.0f / 0.35815147f) * inv_sf * inv_sf);
#pragma unroll
    for (int r = 0; r < 8; ++r) {
        float m = -3.4e38f;
#pragma unroll
        for (int j = 0; j < 13; ++j) m = fmaxf(m, s[j][r]);
        for (int d = 8; d >= 1; d >>= 1) m = fmaxf(m, __shfl_xor(m, d, 32));
#pragma unroll
        for (int j = 0; j < 13; ++j) {
            float xi = (s[j][r] - m) * inv_sf;
            xi = fmaxf(xi, 30.0f * x0i);
            const float q = floorf(xi / x0i);
            const float rr = xi - x0i * q;
            float e = floorf(ldexpf(rr * (rr + bco) + cco, 30 - (int)q));
            s[j][r] = fmaxf(e, 0.0f);
        }
        if (mn >= 5) s[12][r] = 0.0f;
        float sum = 0.0f;
#pragma unroll
        for (int j = 0; j < 13; ++j) sum += s[j][r];
        for (int d = 8; d >= 1; d >>= 1) sum += __shfl_xor(sum, d, 32);
        const float factor = floorf(4294967296.0f / sum);
        const int row = r + 8 * half;
#pragma unroll
        for (int j = 0; j < 13; ++j) {
            const float pi = floorf(s[j][r] * factor * (1.0f / 16777216.0f));
            probs[wave][row][j * 16 + mn] = (_Float16)(pi * (1.0f / 256.0f));
        }
        probs[wave][row][208 + mn] = (_Float16)0.0f;
    }
    __syncthreads();

    v8f acc[4] = {v8f{}, v8f{}, v8f{}, v8f{}};
#pragma unroll
    for (int kt = 0; kt < 7; ++kt) {
        const v16h a = load_frag16(&probs[wave][mn][kt * 32 + 8 * half]);
#pragma unroll
        for (int nt = 0; nt < 4; ++nt) {
            const v16h bf =
                load_frag16(&vtile[nt * 16 + mn][kt * 32 + 8 * half]);
            acc[nt] = __builtin_amdgcn_wmma_f32_16x16x32_f16(
                false, a, false, bf, (short)0, acc[nt], false, false);
        }
    }
    if (active) {
#pragma unroll
        for (int nt = 0; nt < 4; ++nt)
#pragma unroll
            for (int r = 0; r < 8; ++r) {
                const int qr = tile * 16 + r + 8 * half;
                if (qr < S_LEN)
                    out[(baseRow + qr) * D_DIM + headOff + nt * 16 + mn] =
                        acc[nt][r];
            }
    }
}

extern "C" void kernel_launch(void* const* d_in, const int* in_sizes, int n_in,
                              void* d_out, int out_size, void* d_ws, size_t ws_size,
                              hipStream_t stream) {
    const float* x  = (const float*)d_in[0];
    const float* Wq = (const float*)d_in[1];
    const float* bq = (const float*)d_in[2];
    const float* Wk = (const float*)d_in[3];
    const float* bk = (const float*)d_in[4];
    const float* Wv = (const float*)d_in[5];
    const float* bv = (const float*)d_in[6];
    float* out = (float*)d_out;

    constexpr size_t N_X    = (size_t)ROWS * D_DIM;
    constexpr size_t VT_SZ  = (size_t)B_DIM * H_NUM * HD * SPAD * 2;
    constexpr size_t X8_OFF = 256;
    constexpr size_t W8_OFF = X8_OFF + N_X;
    constexpr size_t BS_OFF = W8_OFF + 3ull * D_DIM * D_DIM;
    constexpr size_t BI_OFF = BS_OFF + 3ull * D_DIM * sizeof(float);
    constexpr size_t Q_OFF  = BI_OFF + 3ull * D_DIM * sizeof(float);
    constexpr size_t K_OFF  = Q_OFF + N_X * sizeof(_Float16);
    constexpr size_t VT_OFF = K_OFF + N_X * sizeof(_Float16);

    char* ws = (char*)d_ws;
    unsigned int* d_amax = (unsigned int*)ws;
    int8_t*   d_x8  = (int8_t*)(ws + X8_OFF);
    int8_t*   d_w8  = (int8_t*)(ws + W8_OFF);
    float*    d_bs  = (float*)(ws + BS_OFF);
    float*    d_bi  = (float*)(ws + BI_OFF);
    _Float16* d_Q   = (_Float16*)(ws + Q_OFF);
    _Float16* d_K   = (_Float16*)(ws + K_OFF);
    _Float16* d_Vt  = (_Float16*)(ws + VT_OFF);

    (void)hipMemsetAsync(d_amax, 0, sizeof(unsigned int), stream);
    (void)hipMemsetAsync(d_Vt, 0, VT_SZ, stream);

    absmax_kernel<<<2048, 256, 0, stream>>>(x, N_X, d_amax);

    quant_w_kernel<<<(3 * D_DIM + 255) / 256, 256, 0, stream>>>(
        Wq, bq, Wk, bk, Wv, bv, (const float*)d_amax, d_w8, d_bs, d_bi);

    quant_x_kernel<<<4096, 256, 0, stream>>>(x, (const float*)d_amax, d_x8, N_X);

    qkv_wmma_kernel<<<(ROWT * CGRP * 3) / 4, 128, 0, stream>>>(
        d_x8, d_w8, d_bs, d_bi, d_Q, d_K, d_Vt);

    attn_wmma_kernel<<<B_DIM * H_NUM * 4, 128, 0, stream>>>(
        d_Q, d_K, d_Vt, out, (const float*)d_amax);
}
